// MultiHeadAttention_53077205844362
// MI455X (gfx1250) — matrix operand order repaired and checked by reading, not compiled, not run on hardware
//
#include <hip/hip_runtime.h>

// ---------------------------------------------------------------------------
// MHA forward for MI455X (gfx1250, wave32, WMMA).
// bf16 WMMA (16x16x32) everywhere, fp32 accumulate. Intermediates in d_ws
// (64 MB) stay resident in the 192 MB L2.
// ---------------------------------------------------------------------------

#define S_LEN  2048
#define DMODEL 1024
#define NHEAD  16
#define HDIM   64
#define SP     (S_LEN + 4)   // padded LDS row stride (floats) -> conflict-free

typedef __attribute__((ext_vector_type(16))) __bf16 v16bf;
typedef __attribute__((ext_vector_type(8)))  float  v8f;

union FragU {
    v16bf        v;
    uint4        q[2];
    unsigned int u[8];
};

// fp32 -> bf16, round-to-nearest-even
__device__ __forceinline__ unsigned short f2bf(float f) {
    unsigned int u = __builtin_bit_cast(unsigned int, f);
    u += 0x7FFFu + ((u >> 16) & 1u);
    return (unsigned short)(u >> 16);
}

// A-fragment (16x32 bf16, MxK): lane = row%16 | kg<<4.
// elems 0..7 = K[k+kg*8 .. +8), elems 8..15 = K[k+16+kg*8 .. +8)
__device__ __forceinline__ void load_a(const unsigned short* __restrict__ base,
                                       int ld, int row, int kg, int k, FragU& f) {
    const unsigned short* p = base + row * ld + k + kg * 8;
    f.q[0] = *(const uint4*)(p);
    f.q[1] = *(const uint4*)(p + 16);
}

// B-fragment (32x16 bf16, KxN): lane = col%16 | kg<<4.
// elems 0..7 = K[k+kg*8 .. +8), elems 8..15 = K[k+16+kg*8 .. +8)
__device__ __forceinline__ void load_b(const unsigned short* __restrict__ base,
                                       int ld, int col, int kg, int k, FragU& f) {
    const unsigned short* p = base + col * ld + k + kg * 8;
    f.q[0] = *(const uint4*)(p);
    f.q[1] = *(const uint4*)(p + 16);
}

__device__ __forceinline__ v8f wmma_bf16(const FragU& a, const FragU& b, v8f c) {
    return __builtin_amdgcn_wmma_f32_16x16x32_bf16(
        false, a.v, false, b.v, (short)0, c, false, false);
}

// ---------------------------------------------------------------------------
// fp32 -> bf16 conversion, 8 elements per thread (two b128 loads, one b128 st)
// ---------------------------------------------------------------------------
__global__ void cvt_f32_bf16(const float* __restrict__ in,
                             unsigned short* __restrict__ out, int n8) {
    int i = blockIdx.x * blockDim.x + threadIdx.x;
    if (i >= n8) return;
    const float4* src = (const float4*)(in) + 2 * (size_t)i;
    float4 a = src[0], b = src[1];
    uint4 r;
    r.x = (unsigned)f2bf(a.x) | ((unsigned)f2bf(a.y) << 16);
    r.y = (unsigned)f2bf(a.z) | ((unsigned)f2bf(a.w) << 16);
    r.z = (unsigned)f2bf(b.x) | ((unsigned)f2bf(b.y) << 16);
    r.w = (unsigned)f2bf(b.z) | ((unsigned)f2bf(b.w) << 16);
    ((uint4*)out)[i] = r;
}

// ---------------------------------------------------------------------------
// Y = X @ W^T + bias.  X:[4096,1024] bf16, W:[1024,1024] bf16, both row-major
// (row n of W == column n of W^T -> both operands K-contiguous).
// Per-wave tile: 32 rows x 64 cols = 8 WMMAs per k-step. All 6 fragments of a
// k-step are loaded into distinct registers BEFORE the WMMA burst so the
// scheduler can clause the loads and stagger s_wait_loadcnt.
// mode 0: bf16 out, head-split  [B,H,S,hd]
// mode 1: bf16 out, head-splitT [B,H,hd,S]   (for V)
// mode 2: f32  out, row-major   [B*S, D]     (final output)
// ---------------------------------------------------------------------------
__global__ __launch_bounds__(256)
void gemm_xwT(const unsigned short* __restrict__ A,
              const unsigned short* __restrict__ W,
              const float* __restrict__ bias,
              void* __restrict__ out, int mode) {
    const int lane = threadIdx.x & 31;
    const int wave = threadIdx.x >> 5;
    const int ln = lane & 15, kg = lane >> 4;
    const int gw = blockIdx.x * 8 + wave;       // global wave id (2048 total)
    const int m0 = (gw >> 4) * 32;              // 128 row tiles of 32
    const int n0 = (gw & 15) * 64;              // 16 col tiles of 64

    v8f acc[2][4] = {};
    const unsigned short* arow0 = A + (size_t)(m0 + ln) * DMODEL;
    const unsigned short* arow1 = A + (size_t)(m0 + 16 + ln) * DMODEL;
    for (int k = 0; k < DMODEL; k += 32) {
        // Speculative prefetch of the next-next k-step.
        __builtin_prefetch(arow0 + k + 64, 0, 1);
        __builtin_prefetch(arow1 + k + 64, 0, 1);

        FragU a0, a1, b[4];
        load_a(A, DMODEL, m0 + ln,      kg, k, a0);
        load_a(A, DMODEL, m0 + 16 + ln, kg, k, a1);
#pragma unroll
        for (int t = 0; t < 4; ++t)
            load_b(W, DMODEL, n0 + t * 16 + ln, kg, k, b[t]);
#pragma unroll
        for (int t = 0; t < 4; ++t) {
            acc[0][t] = wmma_bf16(a0, b[t], acc[0][t]);
            acc[1][t] = wmma_bf16(a1, b[t], acc[1][t]);
        }
    }

#pragma unroll
    for (int i = 0; i < 2; ++i) {
#pragma unroll
        for (int t = 0; t < 4; ++t) {
#pragma unroll
            for (int r = 0; r < 8; ++r) {
                const int mm = m0 + i * 16 + kg * 8 + r;  // M = r + 8*(lane/16)
                const int nn = n0 + t * 16 + ln;          // N = lane%16
                const float val = acc[i][t][r] + bias[nn];
                const int bb = mm >> 11, ss = mm & (S_LEN - 1);
                const int hh = nn >> 6,  dd = nn & (HDIM - 1);
                if (mode == 0) {
                    ((unsigned short*)out)[(((size_t)(bb * NHEAD + hh) * S_LEN) + ss) * HDIM + dd] = f2bf(val);
                } else if (mode == 1) {
                    ((unsigned short*)out)[(((size_t)(bb * NHEAD + hh) * HDIM) + dd) * S_LEN + ss] = f2bf(val);
                } else {
                    ((float*)out)[(size_t)mm * DMODEL + nn] = val;
                }
            }
        }
    }
}

// ---------------------------------------------------------------------------
// Attention: one block = 16 query rows of one (b,h).
//  Phase 1: scores[16 x 2048] = (Q K^T)/8 via WMMA  -> LDS fp32 (8 waves)
//  Phase 2: row softmax (shfl_xor reductions)       -> LDS in place
//  Phase 3: ctx[16 x 64] = P @ V, K split across wave pairs (all 8 waves),
//           partials combined through LDS.
// ---------------------------------------------------------------------------
__global__ __launch_bounds__(256)
void attn_kernel(const unsigned short* __restrict__ qh,
                 const unsigned short* __restrict__ kh,
                 const unsigned short* __restrict__ vht,
                 unsigned short* __restrict__ ctx) {
    __shared__ float sc[16 * SP];        // 131,328 B score/P panel
    __shared__ float part[4][16][16];    // 4 KB PV partials (upper wave half)

    const int lane = threadIdx.x & 31;
    const int wave = threadIdx.x >> 5;
    const int ln = lane & 15, kg = lane >> 4;
    const int s0 = blockIdx.x * 16;
    const int bh = blockIdx.y;
    const int b = bh >> 4, h = bh & 15;

    const unsigned short* qbase = qh  + (size_t)(bh * S_LEN + s0) * HDIM;
    const unsigned short* kbase = kh  + (size_t)bh * S_LEN * HDIM;
    const unsigned short* vbase = vht + (size_t)bh * HDIM * S_LEN;

    // ---- Phase 1: QK^T (K=64 -> 2 wmma/tile); wave w covers 256 key cols
    FragU aq0, aq1;
    load_a(qbase, HDIM, ln, kg, 0,  aq0);
    load_a(qbase, HDIM, ln, kg, 32, aq1);
    for (int t = 0; t < 16; ++t) {
        const int n0 = wave * 256 + t * 16;
        v8f acc = {};
        FragU bk0, bk1;
        load_b(kbase, HDIM, n0 + ln, kg, 0,  bk0);
        load_b(kbase, HDIM, n0 + ln, kg, 32, bk1);
        acc = wmma_bf16(aq0, bk0, acc);
        acc = wmma_bf16(aq1, bk1, acc);
#pragma unroll
        for (int r = 0; r < 8; ++r)
            sc[(kg * 8 + r) * SP + n0 + ln] = acc[r] * 0.125f;  // 1/sqrt(64)
    }
    __syncthreads();

    // ---- Phase 2: softmax, 2 rows per wave
#pragma unroll
    for (int rr = 0; rr < 2; ++rr) {
        float* p = sc + (wave * 2 + rr) * SP;
        float mx = -3.402823466e38f;
        for (int i = lane; i < S_LEN; i += 32) mx = fmaxf(mx, p[i]);
        for (int off = 16; off > 0; off >>= 1) mx = fmaxf(mx, __shfl_xor(mx, off, 32));
        float sum = 0.f;
        for (int i = lane; i < S_LEN; i += 32) { float e = __expf(p[i] - mx); p[i] = e; sum += e; }
        for (int off = 16; off > 0; off >>= 1) sum += __shfl_xor(sum, off, 32);
        const float inv = 1.f / sum;
        for (int i = lane; i < S_LEN; i += 32) p[i] *= inv;
    }
    __syncthreads();

    // ---- Phase 3: P @ V. wave -> (d-tile = wave&3, k-half = wave>>2)
    const int d0 = (wave & 3) * 16;
    const int kbeg = (wave >> 2) * (S_LEN / 2);
    v8f acc = {};
    for (int k = kbeg; k < kbeg + S_LEN / 2; k += 32) {
        FragU bv;  load_b(vbase, S_LEN, d0 + ln, kg, k, bv);
        FragU a;                          // A-frag from LDS fp32 -> bf16
        const float* c0 = sc + ln * SP + k + kg * 8;
        const float* c1 = c0 + 16;
#pragma unroll
        for (int j = 0; j < 4; ++j) {
            a.u[j]     = (unsigned)f2bf(c0[2 * j]) | ((unsigned)f2bf(c0[2 * j + 1]) << 16);
            a.u[4 + j] = (unsigned)f2bf(c1[2 * j]) | ((unsigned)f2bf(c1[2 * j + 1]) << 16);
        }
        acc = wmma_bf16(a, bv, acc);
    }
    if (wave >= 4) {                      // upper k-half deposits partials
#pragma unroll
        for (int r = 0; r < 8; ++r) part[wave - 4][kg * 8 + r][ln] = acc[r];
    }
    __syncthreads();
    if (wave < 4) {                       // lower k-half combines + stores
#pragma unroll
        for (int r = 0; r < 8; ++r) {
            const float val = acc[r] + part[wave][kg * 8 + r][ln];
            const int m   = s0 + kg * 8 + r;
            const int col = h * HDIM + d0 + ln;
            ctx[(size_t)(b * S_LEN + m) * DMODEL + col] = f2bf(val);
        }
    }
}

// ---------------------------------------------------------------------------
// Host-side orchestration (all on `stream`; workspace layout below, 64 MB)
// ---------------------------------------------------------------------------
extern "C" void kernel_launch(void* const* d_in, const int* in_sizes, int n_in,
                              void* d_out, int out_size, void* d_ws, size_t ws_size,
                              hipStream_t stream) {
    const float* q   = (const float*)d_in[0];
    const float* k   = (const float*)d_in[1];
    const float* v   = (const float*)d_in[2];
    const float* w_q = (const float*)d_in[3];
    const float* b_q = (const float*)d_in[4];
    const float* w_k = (const float*)d_in[5];
    const float* b_k = (const float*)d_in[6];
    const float* w_v = (const float*)d_in[7];
    const float* b_v = (const float*)d_in[8];
    const float* w_o = (const float*)d_in[9];
    const float* b_o = (const float*)d_in[10];

    char* ws = (char*)d_ws;
    const size_t MB = 1u << 20;
    unsigned short* q_bf  = (unsigned short*)(ws +  0 * MB);  // [4096,1024]
    unsigned short* k_bf  = (unsigned short*)(ws +  8 * MB);
    unsigned short* v_bf  = (unsigned short*)(ws + 16 * MB);
    unsigned short* wq_bf = (unsigned short*)(ws + 24 * MB);  // [1024,1024]
    unsigned short* wk_bf = (unsigned short*)(ws + 26 * MB);
    unsigned short* wv_bf = (unsigned short*)(ws + 28 * MB);
    unsigned short* wo_bf = (unsigned short*)(ws + 30 * MB);
    unsigned short* qh    = (unsigned short*)(ws + 32 * MB);  // [B,H,S,hd]
    unsigned short* kh    = (unsigned short*)(ws + 40 * MB);  // [B,H,S,hd]
    unsigned short* vht   = (unsigned short*)(ws + 48 * MB);  // [B,H,hd,S]
    unsigned short* ctxb  = (unsigned short*)(ws + 56 * MB);  // [4096,1024]

    const int nA8 = 2 * S_LEN * DMODEL / 8;     // 524,288 threads of 8 elems
    const int nW8 = DMODEL * DMODEL / 8;        // 131,072
    cvt_f32_bf16<<<(nA8 + 255) / 256, 256, 0, stream>>>(q,   q_bf,  nA8);
    cvt_f32_bf16<<<(nA8 + 255) / 256, 256, 0, stream>>>(k,   k_bf,  nA8);
    cvt_f32_bf16<<<(nA8 + 255) / 256, 256, 0, stream>>>(v,   v_bf,  nA8);
    cvt_f32_bf16<<<(nW8 + 255) / 256, 256, 0, stream>>>(w_q, wq_bf, nW8);
    cvt_f32_bf16<<<(nW8 + 255) / 256, 256, 0, stream>>>(w_k, wk_bf, nW8);
    cvt_f32_bf16<<<(nW8 + 255) / 256, 256, 0, stream>>>(w_v, wv_bf, nW8);
    cvt_f32_bf16<<<(nW8 + 255) / 256, 256, 0, stream>>>(w_o, wo_bf, nW8);

    // 2048 wave-tiles (128 m-tiles of 32 x 16 n-tiles of 64), 8 waves/block
    gemm_xwT<<<256, 256, 0, stream>>>(q_bf, wq_bf, b_q, qh,  0);
    gemm_xwT<<<256, 256, 0, stream>>>(k_bf, wk_bf, b_k, kh,  0);
    gemm_xwT<<<256, 256, 0, stream>>>(v_bf, wv_bf, b_v, vht, 1);

    attn_kernel<<<dim3(S_LEN / 16, 2 * NHEAD), 256, 0, stream>>>(qh, kh, vht, ctxb);

    gemm_xwT<<<256, 256, 0, stream>>>(ctxb, wo_bf, b_o, d_out, 2);
}
